// IFSLayer_40802189312791
// MI455X (gfx1250) — compile-verified, matrix operand order checked by reading, not run on hardware
//
#include <hip/hip_runtime.h>
#include <hip/hip_bf16.h>
#include <stdint.h>

// Problem constants (from reference)
#define N_OBS  65536
#define N_DIM  256
#define N_FUNC 16
#define N_APP  8

// Tiling
#define TM       256          // rows per workgroup (state resident in LDS)
#define XSTRIDE  260          // f32 stride (pad +4 to break LDS bank conflicts)
#define XROWS    (TM + 16)    // + dummy rows for bucket padding
#define NTHREADS 256
#define NWAVES   8
#define MAXGRP   32           // max padded 16-row groups per step (<= 31)

typedef __attribute__((ext_vector_type(16))) __bf16          v16bf;
typedef __attribute__((ext_vector_type(8)))  float           v8f;
typedef __attribute__((ext_vector_type(4)))  float           f32x4;

__device__ __forceinline__ unsigned short f2bf_bits(float f) {
  union { float f; unsigned u; } cv; cv.f = f;
  unsigned u = cv.u;
  // round-to-nearest-even fp32 -> bf16 (prologue only)
  unsigned r = (u + 0x7fffu + ((u >> 16) & 1u)) >> 16;
  return (unsigned short)r;
}

// ---------------------------------------------------------------------------
// Prologue: swizzle W (f32, [f][k][n]) into bf16 WMMA B-fragment layout.
// Fragment (f, nb, kb) = 512 halves, lane-major: dst[lane*16 + e] holds
// W[f][ kb*32 + hi*8 + (e<8 ? e : e+8) ][ nb*16 + (lane&15) ], hi = lane>=16.
// Each lane then loads its 16 halves with two contiguous b128 loads.
// ---------------------------------------------------------------------------
__global__ void ifs_swizzleW_kernel(const float* __restrict__ W,
                                    unsigned short* __restrict__ Wsw) {
  int gid  = blockIdx.x * blockDim.x + threadIdx.x;   // 16*16*8*32*16 = 1048576
  int e    = gid & 15;
  int lane = (gid >> 4) & 31;
  int kb   = (gid >> 9) & 7;
  int nb   = (gid >> 12) & 15;
  int f    = gid >> 16;
  int hi   = lane >> 4;
  int n    = nb * 16 + (lane & 15);
  int k    = kb * 32 + hi * 8 + (e < 8 ? e : e + 8);
  Wsw[gid] = f2bf_bits(W[((size_t)f * N_DIM + k) * N_DIM + n]);
}

// ---------------------------------------------------------------------------
// Main fused kernel: 256 rows resident in LDS (f32) across all 8 steps.
// Per step: LDS counting-sort rows by function index into padded 16-row
// groups, then dense bf16 WMMA per group against the selected W, with
// explicit double-buffered B-fragment prefetch. Bias lives in LDS so its
// reads ride DScnt and never drain the VMEM (loadcnt) prefetch pipeline.
// ---------------------------------------------------------------------------
__global__ __launch_bounds__(NTHREADS) void ifs_kernel(
    const float* __restrict__ x,
    const int*   __restrict__ fidx,
    const float* __restrict__ bias,
    const unsigned short* __restrict__ Wsw,
    float* __restrict__ out)
{
  extern __shared__ char smem_raw[];
  float* X       = (float*)smem_raw;                                  // XROWS*XSTRIDE f32
  int*   rowlist = (int*)(smem_raw + (size_t)XROWS * XSTRIDE * 4);    // 512
  int*   cnt     = rowlist + 512;                                     // 16
  int*   off     = cnt + N_FUNC;                                      // 17
  int*   grpF    = off + N_FUNC + 1;                                  // 32
  int*   meta    = grpF + MAXGRP;                                     // 1
  float* biasLds = (float*)(meta + 1);                                // 4096 f32 (16 KB)

  const int t        = threadIdx.x;
  const int row_base = blockIdx.x * TM;
  const int wave     = t >> 5;
  const int lane     = t & 31;
  const int lo16     = lane & 15;
  const int hi8      = (lane >> 4) << 3;   // 0 or 8

  // ---- stage bias table into LDS (16 KB; reads then use DScnt only) ----
  {
    const f32x4* bsrc = (const f32x4*)bias;
    #pragma unroll
    for (int j = 0; j < (N_FUNC * N_DIM / 4) / NTHREADS; ++j) {
      int q = t + j * NTHREADS;
      *(f32x4*)&biasLds[q * 4] = bsrc[q];
    }
  }

  // ---- load x tile into LDS state (f32, coalesced b128) ----
  {
    const f32x4* src = (const f32x4*)(x + (size_t)row_base * N_DIM);
    for (int j = 0; j < (TM * N_DIM / 4) / NTHREADS; ++j) {
      int q   = t + j * NTHREADS;
      f32x4 v = src[q];
      int row = (q * 4) / N_DIM;
      int col = (q * 4) % N_DIM;
      *(f32x4*)&X[row * XSTRIDE + col] = v;
    }
  }

  for (int s = 0; s < N_APP; ++s) {
    __syncthreads();                       // prev step's writes visible
    // ---- bucket rows by function index ----
    rowlist[t]            = TM + (t & 15); // pre-fill with dummy row ids
    rowlist[t + NTHREADS] = TM + (t & 15);
    if (t < N_FUNC) cnt[t] = 0;
    __syncthreads();
    const int myf   = fidx[(size_t)(row_base + t) * N_APP + s];
    const int mypos = atomicAdd(&cnt[myf], 1);
    __syncthreads();
    if (t == 0) {                          // serial prefix scan w/ 16-padding
      int o = 0;
      for (int f = 0; f < N_FUNC; ++f) {
        off[f] = o;
        int pg = (cnt[f] + 15) >> 4;
        for (int gi = 0; gi < pg; ++gi) grpF[(o >> 4) + gi] = f;
        o += pg << 4;
      }
      off[N_FUNC] = o;
      meta[0] = o >> 4;
    }
    __syncthreads();
    rowlist[off[myf] + mypos] = t;
    __syncthreads();

    // ---- compute: per-wave, whole 16-row groups (EXEC stays all-ones) ----
    const int ngroups = meta[0];
    for (int g = wave; g < ngroups; g += NWAVES) {
      const int  f     = grpF[g];
      const int* rl    = rowlist + g * 16;
      const int  myrow = rl[lo16];
      int wr[8];
      #pragma unroll
      for (int v = 0; v < 8; ++v) wr[v] = rl[v + hi8];   // write-back rows

      // A fragments: gather this wave's 16 rows from f32 state, HW cvt->bf16.
      // 16-bit A layout: lane<16 holds K {0..7,16..23}, lane>=16 {8..15,24..31}.
      v16bf a[8];
      #pragma unroll
      for (int kb = 0; kb < 8; ++kb) {
        const float* p = &X[myrow * XSTRIDE + kb * 32 + hi8];
        f32x4 q0 = *(const f32x4*)(p);
        f32x4 q1 = *(const f32x4*)(p + 4);
        f32x4 q2 = *(const f32x4*)(p + 16);
        f32x4 q3 = *(const f32x4*)(p + 20);
        v16bf va;
        #pragma unroll
        for (int j = 0; j < 4; ++j) {
          va[j]      = (__bf16)q0[j];
          va[j + 4]  = (__bf16)q1[j];
          va[j + 8]  = (__bf16)q2[j];
          va[j + 12] = (__bf16)q3[j];
        }
        a[kb] = va;
      }

      const unsigned short* wbase = Wsw + ((size_t)f * 16 * 8) * 512 + lane * 16;
      const float* bbase = biasLds + f * N_DIM + lo16;

      // Double-buffered B-fragment pipeline: while the WMMA chain consumes
      // one buffer, the other buffer's 16 b128 loads are in flight.
      v16bf b0[8], b1[8];
      #pragma unroll
      for (int kb = 0; kb < 8; ++kb)
        b0[kb] = *(const v16bf*)(wbase + (size_t)kb * 512);

      #pragma unroll 1
      for (int nbp = 0; nbp < 8; ++nbp) {
        const int nb0 = nbp * 2;
        const int nb1 = nb0 + 1;

        // prefetch odd tile
        const unsigned short* w1 = wbase + (size_t)nb1 * 4096;
        #pragma unroll
        for (int kb = 0; kb < 8; ++kb)
          b1[kb] = *(const v16bf*)(w1 + (size_t)kb * 512);

        // compute even tile (C/D layout: VGPR v -> row v+hi8, col lo16)
        {
          float bv = bbase[nb0 * 16];      // LDS read (DScnt, not loadcnt)
          v8f c;
          #pragma unroll
          for (int v = 0; v < 8; ++v) c[v] = bv;
          #pragma unroll
          for (int kb = 0; kb < 8; ++kb)
            c = __builtin_amdgcn_wmma_f32_16x16x32_bf16(
                  false, a[kb], false, b0[kb], (short)0, c, false, false);
          #pragma unroll
          for (int v = 0; v < 8; ++v)
            X[wr[v] * XSTRIDE + nb0 * 16 + lo16] = c[v];
        }

        // prefetch next even tile (uniform branch; keeps pipeline primed)
        if (nbp < 7) {
          const unsigned short* w2 = wbase + (size_t)(nb0 + 2) * 4096;
          #pragma unroll
          for (int kb = 0; kb < 8; ++kb)
            b0[kb] = *(const v16bf*)(w2 + (size_t)kb * 512);
        }

        // compute odd tile
        {
          float bv = bbase[nb1 * 16];      // LDS read (DScnt, not loadcnt)
          v8f c;
          #pragma unroll
          for (int v = 0; v < 8; ++v) c[v] = bv;
          #pragma unroll
          for (int kb = 0; kb < 8; ++kb)
            c = __builtin_amdgcn_wmma_f32_16x16x32_bf16(
                  false, a[kb], false, b1[kb], (short)0, c, false, false);
          #pragma unroll
          for (int v = 0; v < 8; ++v)
            X[wr[v] * XSTRIDE + nb1 * 16 + lo16] = c[v];
        }
      }
    }
  }

  __syncthreads();
  // ---- store final state ----
  {
    f32x4* dst = (f32x4*)(out + (size_t)row_base * N_DIM);
    for (int j = 0; j < (TM * N_DIM / 4) / NTHREADS; ++j) {
      int q   = t + j * NTHREADS;
      int row = (q * 4) / N_DIM;
      int col = (q * 4) % N_DIM;
      dst[q] = *(const f32x4*)&X[row * XSTRIDE + col];
    }
  }
}

extern "C" void kernel_launch(void* const* d_in, const int* in_sizes, int n_in,
                              void* d_out, int out_size, void* d_ws, size_t ws_size,
                              hipStream_t stream) {
  (void)in_sizes; (void)n_in; (void)out_size; (void)ws_size;
  const float* x    = (const float*)d_in[0];
  const int*   fi   = (const int*)d_in[1];
  const float* W    = (const float*)d_in[2];
  const float* b    = (const float*)d_in[3];
  unsigned short* Wsw = (unsigned short*)d_ws;   // 2 MB bf16 swizzled W

  // Prologue: W f32 -> bf16 WMMA-fragment layout (L2-resident afterwards)
  ifs_swizzleW_kernel<<<(N_FUNC * N_DIM * N_DIM) / 256, 256, 0, stream>>>(W, Wsw);

  constexpr size_t SMEM = (size_t)XROWS * XSTRIDE * 4   // state
                        + 512 * 4                       // rowlist
                        + N_FUNC * 4                    // cnt
                        + (N_FUNC + 1) * 4              // off
                        + MAXGRP * 4                    // grpF
                        + 4                             // meta
                        + (size_t)N_FUNC * N_DIM * 4;   // bias table in LDS
  hipFuncSetAttribute((const void*)ifs_kernel,
                      hipFuncAttributeMaxDynamicSharedMemorySize, (int)SMEM);
  ifs_kernel<<<N_OBS / TM, NTHREADS, SMEM, stream>>>(x, fi, b, Wsw, (float*)d_out);
}
